// TitansTransformer_51110110822852
// MI455X (gfx1250) — matrix operand order repaired and checked by reading, not compiled, not run on hardware
//
#include <hip/hip_runtime.h>
#include <math.h>

#define Dm     1024
#define DEPTHc 4
#define Hh     16
#define DHd    64
#define MLPd   4096
#define DKd    64
#define DVd    64
#define Bb     4
#define Ll     2048
#define ALPHAc 0.02f
#define LRc    0.1f
#define BETAc  0.9f

typedef __attribute__((ext_vector_type(16))) __bf16 v16bf;
typedef __attribute__((ext_vector_type(8)))  float  v8f;

union ABfrag { v16bf v; uint4 q[2]; unsigned short u[16]; };

__device__ __forceinline__ unsigned short f2bf(float f) {
    unsigned int u = __float_as_uint(f);
    u += 0x7FFFu + ((u >> 16) & 1u);   // round-to-nearest-even
    return (unsigned short)(u >> 16);
}

__device__ __forceinline__ float gelu_tanh(float x) {
    float x3 = x * x * x;
    return 0.5f * x * (1.0f + tanhf(0.7978845608028654f * (x + 0.044715f * x3)));
}

// LDS byte address for async Global->LDS ops: per ISA 10.2, addresses in the LDS
// aperture map to LDS by truncation to the low 32 bits.
__device__ __forceinline__ unsigned lds_addr32(const void* p) {
    return (unsigned)(unsigned long long)p;
}

// ---------------------------------------------------------------------------
// bf16-WMMA GEMM: C[M,N] = act( A[M,K] @ B[K,N] + bias + resScale*Res )
// Requirements (hold for every call in this workload): M % 128 == 0, K % 64 == 0.
// NG=true adds column guards for N < 128-tile coverage (used only for N=64).
// Block tile 128x128, K-step 64, 8 waves each computing 32x64 (16 WMMA / tile).
// ---------------------------------------------------------------------------
template <bool NG>
__global__ __launch_bounds__(256) void gemm_bf16_wmma(
    const float* __restrict__ A, const float* __restrict__ Bm,
    const float* __restrict__ bias, const float* __restrict__ Res,
    float* __restrict__ C, int M, int N, int K,
    long long sA, long long sB, long long sC, long long sR,
    float resScale, int act)
{
    __shared__ __align__(16) unsigned short As[128][72];  // [row][k], padded
    __shared__ __align__(16) unsigned short Bs[128][72];  // [col][k] (B transposed)

    const int tid  = threadIdx.x;
    const int lane = tid & 31;
    const int wave = tid >> 5;
    const int wm   = wave >> 1;          // 0..3  -> 32 rows
    const int wn   = wave & 1;           // 0..1  -> 64 cols
    const int m0   = blockIdx.y * 128;
    const int n0   = blockIdx.x * 128;
    const int z    = blockIdx.z;

    A  += (long long)z * sA;
    Bm += (long long)z * sB;
    C  += (long long)z * sC;
    if (Res) Res += (long long)z * sR;

    v8f vzero = {};
    v8f acc[2][4];
#pragma unroll
    for (int i = 0; i < 2; ++i)
#pragma unroll
        for (int j = 0; j < 4; ++j) acc[i][j] = vzero;

    for (int kt = 0; kt < K; kt += 64) {
        __syncthreads();
        // ---- stage A tile (128x64 fp32 -> bf16); no guards (M%128==0, K%64==0)
#pragma unroll
        for (int i = 0; i < 8; ++i) {
            int c  = tid + i * 256;       // 2048 float4 chunks
            int r  = c >> 4;              // 16 float4 per row
            int kp = (c & 15) << 2;
            float4 f = *(const float4*)(A + (long long)(m0 + r) * K + kt + kp);
            As[r][kp] = f2bf(f.x); As[r][kp + 1] = f2bf(f.y);
            As[r][kp + 2] = f2bf(f.z); As[r][kp + 3] = f2bf(f.w);
        }
        // ---- stage B tile (64x128 fp32 -> bf16, transposed into Bs[col][k]) ----
#pragma unroll
        for (int i = 0; i < 8; ++i) {
            int c  = tid + i * 256;
            int kr = c >> 5;              // 32 float4 per row
            int np = (c & 31) << 2;
            float4 f = make_float4(0.f, 0.f, 0.f, 0.f);
            if (!NG || (n0 + np + 4 <= N))
                f = *(const float4*)(Bm + (long long)(kt + kr) * N + n0 + np);
            Bs[np][kr]     = f2bf(f.x);
            Bs[np + 1][kr] = f2bf(f.y);
            Bs[np + 2][kr] = f2bf(f.z);
            Bs[np + 3][kr] = f2bf(f.w);
        }
        // prefetch next K-tile while WMMAs run
        if (kt + 64 < K) {
            int r  = tid >> 4; int kp = (tid & 15) << 2;
            __builtin_prefetch(A + (long long)(m0 + r) * K + kt + 64 + kp, 0, 0);
            int kr = tid >> 5; int np = (tid & 31) << 2;
            if (!NG || (n0 + np + 4 <= N))
                __builtin_prefetch(Bm + (long long)(kt + 64 + kr) * N + n0 + np, 0, 0);
        }
        __syncthreads();

        // ---- two WMMA K-chunks per staged tile ----
#pragma unroll
        for (int kk = 0; kk < 64; kk += 32) {
            ABfrag af[2], bf[4];
#pragma unroll
            for (int mi = 0; mi < 2; ++mi) {
                int r  = wm * 32 + mi * 16 + (lane & 15);
                int c0 = kk + ((lane >> 4) << 3);   // A: halves K 0..7/16..23 vs 8..15/24..31
                af[mi].q[0] = *(const uint4*)&As[r][c0];
                af[mi].q[1] = *(const uint4*)&As[r][16 + c0];
            }
#pragma unroll
            for (int ni = 0; ni < 4; ++ni) {
                int cb = wn * 64 + ni * 16 + (lane & 15);
                int k0 = kk + ((lane >> 4) << 3);   // B: lane=col, halves K 0..7/16..23 vs 8..15/24..31
                bf[ni].q[0] = *(const uint4*)&Bs[cb][k0];
                bf[ni].q[1] = *(const uint4*)&Bs[cb][k0 + 16];
            }
#pragma unroll
            for (int mi = 0; mi < 2; ++mi)
#pragma unroll
                for (int ni = 0; ni < 4; ++ni)
                    acc[mi][ni] = __builtin_amdgcn_wmma_f32_16x16x32_bf16(
                        false, af[mi].v, false, bf[ni].v, (short)0, acc[mi][ni],
                        false, false);
        }
    }

    // ---- epilogue: C layout (VGPR j: M=j / j+8 per half-wave, N=lane&15) ----
#pragma unroll
    for (int mi = 0; mi < 2; ++mi) {
#pragma unroll
        for (int ni = 0; ni < 4; ++ni) {
            int col = n0 + wn * 64 + ni * 16 + (lane & 15);
            if (NG && col >= N) continue;
#pragma unroll
            for (int j = 0; j < 8; ++j) {
                int row = m0 + wm * 32 + mi * 16 + j + ((lane >> 4) << 3);
                float v = acc[mi][ni][j];
                if (bias) v += bias[col];
                if (Res)  v += resScale * Res[(long long)row * N + col];
                if (act == 1) v = gelu_tanh(v);
                C[(long long)row * N + col] = v;
            }
        }
    }
}

// ---------------------------------------------------------------------------
// Flash attention: one block = one (b,h) head x 128 query rows, 8 waves x 16 rows.
// qkv layout [B,L,3*H*DH]; output O [B,L,H*DH]. Softmax scale folded into q.
// ---------------------------------------------------------------------------
__global__ __launch_bounds__(256) void attn_flash_wmma(
    const float* __restrict__ qkv, float* __restrict__ O)
{
    __shared__ __align__(16) unsigned short Qs[128][72];
    __shared__ __align__(16) unsigned short Ks[32][72];
    __shared__ __align__(16) unsigned short Vt[64][40];      // [d][key]
    __shared__ __align__(16) unsigned short Ps[8][16][40];   // per-wave P tile

    const int tid = threadIdx.x, lane = tid & 31, wave = tid >> 5;
    const int half = lane >> 4;
    const int bh = blockIdx.y, b = bh / Hh, h = bh % Hh;
    const int q0 = blockIdx.x * 128;
    const int ld = 3 * Hh * DHd;
    const long long base = (long long)b * Ll * ld;
    const int offq = h * DHd, offk = Hh * DHd + h * DHd, offv = 2 * Hh * DHd + h * DHd;

    // stage q tile (scaled by DH^-0.5 = 0.125)
#pragma unroll
    for (int i = 0; i < 8; ++i) {
        int c = tid + i * 256;             // 2048 float4 chunks
        int r = c >> 4; int dp = (c & 15) << 2;
        float4 f = *(const float4*)(qkv + base + (long long)(q0 + r) * ld + offq + dp);
        Qs[r][dp]     = f2bf(f.x * 0.125f);
        Qs[r][dp + 1] = f2bf(f.y * 0.125f);
        Qs[r][dp + 2] = f2bf(f.z * 0.125f);
        Qs[r][dp + 3] = f2bf(f.w * 0.125f);
    }
    __syncthreads();

    ABfrag aq[2];
#pragma unroll
    for (int s = 0; s < 2; ++s) {          // d slabs 0..31 / 32..63
        int r  = wave * 16 + (lane & 15);
        int c0 = s * 32 + (half << 3);
        aq[s].q[0] = *(const uint4*)&Qs[r][c0];
        aq[s].q[1] = *(const uint4*)&Qs[r][c0 + 16];
    }

    float mrun[8], lrun[8];
    v8f vzero = {};
    v8f oacc[4];
#pragma unroll
    for (int j = 0; j < 8; ++j) { mrun[j] = -1e30f; lrun[j] = 0.f; }
#pragma unroll
    for (int t = 0; t < 4; ++t) oacc[t] = vzero;

    for (int kb = 0; kb < Ll; kb += 32) {
        __syncthreads();
        // stage K block [32][64]
#pragma unroll
        for (int i = 0; i < 2; ++i) {
            int c = tid + i * 256;
            int r = c >> 4; int dp = (c & 15) << 2;
            float4 f = *(const float4*)(qkv + base + (long long)(kb + r) * ld + offk + dp);
            Ks[r][dp] = f2bf(f.x); Ks[r][dp + 1] = f2bf(f.y);
            Ks[r][dp + 2] = f2bf(f.z); Ks[r][dp + 3] = f2bf(f.w);
        }
        // stage V block transposed [64][32]
#pragma unroll
        for (int i = 0; i < 2; ++i) {
            int c = tid + i * 256;
            int r = c >> 4; int dp = (c & 15) << 2;
            float4 f = *(const float4*)(qkv + base + (long long)(kb + r) * ld + offv + dp);
            Vt[dp][r] = f2bf(f.x); Vt[dp + 1][r] = f2bf(f.y);
            Vt[dp + 2][r] = f2bf(f.z); Vt[dp + 3][r] = f2bf(f.w);
        }
        __syncthreads();

        // S[s] = q @ k^T for two groups of 16 keys
        v8f S[2];
#pragma unroll
        for (int s = 0; s < 2; ++s) {
            ABfrag bk0, bk1;
            int key = s * 16 + (lane & 15);
            int t0  = half << 3;
            bk0.q[0] = *(const uint4*)&Ks[key][t0];
            bk0.q[1] = *(const uint4*)&Ks[key][t0 + 16];
            bk1.q[0] = *(const uint4*)&Ks[key][32 + t0];
            bk1.q[1] = *(const uint4*)&Ks[key][32 + t0 + 16];
            v8f zz = vzero;
            zz = __builtin_amdgcn_wmma_f32_16x16x32_bf16(false, aq[0].v, false, bk0.v,
                                                         (short)0, zz, false, false);
            zz = __builtin_amdgcn_wmma_f32_16x16x32_bf16(false, aq[1].v, false, bk1.v,
                                                         (short)0, zz, false, false);
            S[s] = zz;
        }

        // online softmax; rows of VGPR j live across 16 lanes of each half-wave
#pragma unroll
        for (int j = 0; j < 8; ++j) {
            float a = S[0][j], c2 = S[1][j];
            float mx = fmaxf(a, c2);
#pragma unroll
            for (int off = 8; off >= 1; off >>= 1)
                mx = fmaxf(mx, __shfl_xor(mx, off, 32));
            float mn   = fmaxf(mrun[j], mx);
            float corr = __expf(mrun[j] - mn);
            mrun[j] = mn;
            float p0 = __expf(a - mn);
            float p1 = __expf(c2 - mn);
            float rs = p0 + p1;
#pragma unroll
            for (int off = 8; off >= 1; off >>= 1)
                rs += __shfl_xor(rs, off, 32);
            lrun[j] = lrun[j] * corr + rs;
#pragma unroll
            for (int t = 0; t < 4; ++t) oacc[t][j] *= corr;
            int pr = j + (half << 3);
            Ps[wave][pr][lane & 15]        = f2bf(p0);
            Ps[wave][pr][16 + (lane & 15)] = f2bf(p1);
        }

        // reload P in A layout (same-wave LDS ops are in-order)
        ABfrag ap;
        {
            int r  = lane & 15;
            int c0 = half << 3;
            ap.q[0] = *(const uint4*)&Ps[wave][r][c0];
            ap.q[1] = *(const uint4*)&Ps[wave][r][16 + c0];
        }
        // O += P @ V
#pragma unroll
        for (int t = 0; t < 4; ++t) {
            ABfrag bv;
            int d  = t * 16 + (lane & 15);
            int k0 = half << 3;
            bv.q[0] = *(const uint4*)&Vt[d][k0];
            bv.q[1] = *(const uint4*)&Vt[d][k0 + 16];
            oacc[t] = __builtin_amdgcn_wmma_f32_16x16x32_bf16(
                false, ap.v, false, bv.v, (short)0, oacc[t], false, false);
        }
    }

    // write normalized output
    const long long baseO = (long long)b * Ll * (Hh * DHd);
#pragma unroll
    for (int t = 0; t < 4; ++t) {
        int col = h * DHd + t * 16 + (lane & 15);
#pragma unroll
        for (int j = 0; j < 8; ++j) {
            int row = q0 + wave * 16 + j + (half << 3);
            O[baseO + (long long)row * (Hh * DHd) + col] = oacc[t][j] / lrun[j];
        }
    }
}

// ---------------------------------------------------------------------------
// Row LayerNorm over D=1024 (one block per row, 256 threads x 4 elems).
// ---------------------------------------------------------------------------
__global__ __launch_bounds__(256) void layernorm_row(
    const float* __restrict__ src, float* __restrict__ dst,
    const float* __restrict__ g, const float* __restrict__ bta)
{
    __shared__ float red[256];
    const int row = blockIdx.x, tid = threadIdx.x;
    const float* p = src + (long long)row * Dm;
    float x[4]; float s = 0.f;
#pragma unroll
    for (int i = 0; i < 4; ++i) { x[i] = p[tid + i * 256]; s += x[i]; }
    red[tid] = s; __syncthreads();
    for (int o = 128; o; o >>= 1) { if (tid < o) red[tid] += red[tid + o]; __syncthreads(); }
    float mean = red[0] * (1.0f / Dm);
    __syncthreads();
    float v = 0.f;
#pragma unroll
    for (int i = 0; i < 4; ++i) { float d = x[i] - mean; v += d * d; }
    red[tid] = v; __syncthreads();
    for (int o = 128; o; o >>= 1) { if (tid < o) red[tid] += red[tid + o]; __syncthreads(); }
    float rstd = rsqrtf(red[0] * (1.0f / Dm) + 1e-5f);
    float* q = dst + (long long)row * Dm;
#pragma unroll
    for (int i = 0; i < 4; ++i) {
        int c = tid + i * 256;
        q[c] = (x[i] - mean) * rstd * g[c] + bta[c];
    }
}

// ---------------------------------------------------------------------------
// Titans memory update: grad = 2/(L*DV) * K^T @ err; momentum + decay.
// One block per batch b. Tiles of K/err are staged into LDS with the CDNA5
// async Global->LDS path (GLOBAL_LOAD_ASYNC_TO_LDS_B128, ASYNCcnt) — raw fp32
// copy, no conversion needed, so the DMA path applies directly.
// ---------------------------------------------------------------------------
__global__ __launch_bounds__(256) void mem_update_kern(
    const float* __restrict__ Kb, const float* __restrict__ Eb,
    const float* __restrict__ Wm, const float* __restrict__ Sm,
    float* __restrict__ outW, float* __restrict__ outS)
{
    __shared__ __align__(16) float sk[64][64];   // 256B rows: every 16B chunk aligned
    __shared__ __align__(16) float se[64][64];
    const int b = blockIdx.x, tid = threadIdx.x;
    Kb += (long long)b * Ll * DKd;  Eb += (long long)b * Ll * DVd;
    Wm += b * DKd * DVd;  Sm += b * DKd * DVd;
    outW += b * DKd * DVd; outS += b * DKd * DVd;

    const int i  = tid & 63;            // grad row (DK index)
    const int jb = (tid >> 6) << 4;     // 16-wide column group
    float acc[16];
#pragma unroll
    for (int j = 0; j < 16; ++j) acc[j] = 0.f;

    for (int tb = 0; tb < Ll; tb += 64) {
        __syncthreads();                // previous tile fully consumed
        // async DMA: 64x64 fp32 tile each, 4 x b128 per thread per tile
#pragma unroll
        for (int q2 = 0; q2 < 4; ++q2) {
            int c  = tid + q2 * 256;          // float4 chunk id (0..1023)
            int r  = c >> 4;                  // 16 float4 per 64-float row
            int cc = (c & 15) << 2;
            unsigned           lk = lds_addr32(&sk[r][cc]);
            unsigned long long gk = (unsigned long long)(Kb + (long long)(tb + r) * DKd + cc);
            asm volatile("global_load_async_to_lds_b128 %0, %1, off"
                         :: "v"(lk), "v"(gk) : "memory");
            unsigned           le = lds_addr32(&se[r][cc]);
            unsigned long long ge = (unsigned long long)(Eb + (long long)(tb + r) * DVd + cc);
            asm volatile("global_load_async_to_lds_b128 %0, %1, off"
                         :: "v"(le), "v"(ge) : "memory");
        }
        asm volatile("s_wait_asynccnt 0x0" ::: "memory");
        __syncthreads();

        for (int tt = 0; tt < 64; ++tt) {
            float kv = sk[tt][i];
#pragma unroll
            for (int j = 0; j < 16; ++j) acc[j] += kv * se[tt][jb + j];
        }
    }
    const float gs = 2.0f / (float)(Ll * DVd);
#pragma unroll
    for (int j = 0; j < 16; ++j) {
        int idx = i * DVd + jb + j;
        float snew = BETAc * Sm[idx] - LRc * (gs * acc[j]);
        outS[idx] = snew;
        outW[idx] = (1.0f - ALPHAc) * Wm[idx] + snew;
    }
}

// ---------------------------------------------------------------------------
extern "C" void kernel_launch(void* const* d_in, const int* in_sizes, int n_in,
                              void* d_out, int out_size, void* d_ws, size_t ws_size,
                              hipStream_t stream)
{
    const float* x_in = (const float*)d_in[0];
    const float* memW = (const float*)d_in[1];
    const float* memS = (const float*)d_in[2];
    const float* Wk  = (const float*)d_in[3];
    const float* bk  = (const float*)d_in[4];
    const float* Wv  = (const float*)d_in[5];
    const float* bv  = (const float*)d_in[6];
    const float* Wq  = (const float*)d_in[7];
    const float* bq  = (const float*)d_in[8];
    const float* Uw  = (const float*)d_in[9];
    const float* Ub  = (const float*)d_in[10];
    const float* ln1g = (const float*)d_in[11];
    const float* ln1b = (const float*)d_in[12];
    const float* Wqkv = (const float*)d_in[13];
    const float* Wo  = (const float*)d_in[14];
    const float* bo  = (const float*)d_in[15];
    const float* ln2g = (const float*)d_in[16];
    const float* ln2b = (const float*)d_in[17];
    const float* W1  = (const float*)d_in[18];
    const float* b1  = (const float*)d_in[19];
    const float* W2  = (const float*)d_in[20];
    const float* b2  = (const float*)d_in[21];
    const float* lng = (const float*)d_in[22];
    const float* lnb = (const float*)d_in[23];

    const size_t NBLD  = (size_t)Bb * Ll * Dm;   // 8,388,608
    const size_t NBL64 = (size_t)Bb * Ll * 64;   //   524,288
    const size_t need  = 3 * NBLD + (size_t)Bb * Ll * MLPd + 5 * NBL64;
    if (ws_size < need * sizeof(float)) return;

    float* outX = (float*)d_out;
    float* outW = outX + NBLD;
    float* outS = outW + (size_t)DEPTHc * Bb * DKd * DVd;

    float* ws   = (float*)d_ws;
    float* xcur = ws;  ws += NBLD;
    float* xm   = ws;  ws += NBLD;                      // h, later attention O
    float* xa   = ws;  ws += NBLD;
    float* big  = ws;  ws += (size_t)Bb * Ll * MLPd;    // qkv / MLP hidden
    float* Kb2  = ws;  ws += NBL64;
    float* Vb2  = ws;  ws += NBL64;
    float* Qb2  = ws;  ws += NBL64;
    float* Ym   = ws;  ws += NBL64;
    float* Eb2  = ws;  ws += NBL64;

    hipMemcpyAsync(xcur, x_in, NBLD * sizeof(float), hipMemcpyDeviceToDevice, stream);

    const int M = Bb * Ll;   // 8192
    dim3 blk(256);
    auto G = [](int n, int m, int z) { return dim3((n + 127) / 128, (m + 127) / 128, z); };

    for (int l = 0; l < DEPTHc; ++l) {
        const float* Wk_l = Wk + (size_t)l * Dm * DKd;  const float* bk_l = bk + (size_t)l * DKd;
        const float* Wv_l = Wv + (size_t)l * Dm * DVd;  const float* bv_l = bv + (size_t)l * DVd;
        const float* Wq_l = Wq + (size_t)l * Dm * DKd;  const float* bq_l = bq + (size_t)l * DKd;
        const float* Uw_l = Uw + (size_t)l * DVd * Dm;  const float* Ub_l = Ub + (size_t)l * Dm;
        const float* Wqkv_l = Wqkv + (size_t)l * Dm * 3 * Hh * DHd;
        const float* Wo_l = Wo + (size_t)l * Hh * DHd * Dm; const float* bo_l = bo + (size_t)l * Dm;
        const float* W1_l = W1 + (size_t)l * Dm * MLPd; const float* b1_l = b1 + (size_t)l * MLPd;
        const float* W2_l = W2 + (size_t)l * MLPd * Dm; const float* b2_l = b2 + (size_t)l * Dm;
        const float* ln1g_l = ln1g + (size_t)l * Dm;    const float* ln1b_l = ln1b + (size_t)l * Dm;
        const float* ln2g_l = ln2g + (size_t)l * Dm;    const float* ln2b_l = ln2b + (size_t)l * Dm;
        const float* memW_l = memW + (size_t)l * Bb * DKd * DVd;
        const float* memS_l = memS + (size_t)l * Bb * DKd * DVd;
        float* outW_l = outW + (size_t)l * Bb * DKd * DVd;
        float* outS_l = outS + (size_t)l * Bb * DKd * DVd;

        // K/V/Q projections (N=64 -> guarded variant)
        gemm_bf16_wmma<true><<<G(DKd, M, 1), blk, 0, stream>>>(xcur, Wk_l, bk_l, nullptr, Kb2,
            M, DKd, Dm, 0, 0, 0, 0, 0.f, 0);
        gemm_bf16_wmma<true><<<G(DVd, M, 1), blk, 0, stream>>>(xcur, Wv_l, bv_l, nullptr, Vb2,
            M, DVd, Dm, 0, 0, 0, 0, 0.f, 0);
        gemm_bf16_wmma<true><<<G(DKd, M, 1), blk, 0, stream>>>(xcur, Wq_l, bq_l, nullptr, Qb2,
            M, DKd, Dm, 0, 0, 0, 0, 0.f, 0);
        // memory read: y_mem = Q @ mem_W (batched over B)
        gemm_bf16_wmma<true><<<G(DVd, Ll, Bb), blk, 0, stream>>>(Qb2, memW_l, nullptr, nullptr, Ym,
            Ll, DVd, DKd, (long long)Ll * DKd, (long long)DKd * DVd, (long long)Ll * DVd, 0, 0.f, 0);
        // x_mem = y_mem @ Uw + Ub  (N=1024 -> fast variant)
        gemm_bf16_wmma<false><<<G(Dm, M, 1), blk, 0, stream>>>(Ym, Uw_l, Ub_l, nullptr, xm,
            M, Dm, DVd, 0, 0, 0, 0, 0.f, 0);
        // h = LN1(x_mem)   (in place)
        layernorm_row<<<M, blk, 0, stream>>>(xm, xm, ln1g_l, ln1b_l);
        // qkv = h @ Wqkv
        gemm_bf16_wmma<false><<<G(3 * Hh * DHd, M, 1), blk, 0, stream>>>(xm, Wqkv_l, nullptr,
            nullptr, big, M, 3 * Hh * DHd, Dm, 0, 0, 0, 0, 0.f, 0);
        // attention -> reuse xm as O
        attn_flash_wmma<<<dim3(Ll / 128, Bb * Hh), blk, 0, stream>>>(big, xm);
        // x_att = O @ Wo + bo
        gemm_bf16_wmma<false><<<G(Dm, M, 1), blk, 0, stream>>>(xm, Wo_l, bo_l, nullptr, xa,
            M, Dm, Hh * DHd, 0, 0, 0, 0, 0.f, 0);
        // hf = LN2(x_att)  (in place)
        layernorm_row<<<M, blk, 0, stream>>>(xa, xa, ln2g_l, ln2b_l);
        // MLP: big = gelu(hf @ W1 + b1); xcur = big @ W2 + b2 + xcur
        gemm_bf16_wmma<false><<<G(MLPd, M, 1), blk, 0, stream>>>(xa, W1_l, b1_l, nullptr, big,
            M, MLPd, Dm, 0, 0, 0, 0, 0.f, 1);
        gemm_bf16_wmma<false><<<G(Dm, M, 1), blk, 0, stream>>>(big, W2_l, b2_l, xcur, xcur,
            M, Dm, MLPd, 0, 0, 0, 0, 1.f, 0);
        // err = K @ mem_W - V (batched, N=64 guarded)
        gemm_bf16_wmma<true><<<G(DVd, Ll, Bb), blk, 0, stream>>>(Kb2, memW_l, nullptr, Vb2, Eb2,
            Ll, DVd, DKd, (long long)Ll * DKd, (long long)DKd * DVd,
            (long long)Ll * DVd, (long long)Ll * DVd, -1.f, 0);
        // momentum memory update (async Global->LDS staging)
        mem_update_kern<<<Bb, blk, 0, stream>>>(Kb2, Eb2, memW_l, memS_l, outW_l, outS_l);
    }

    // final LayerNorm to output
    layernorm_row<<<M, blk, 0, stream>>>(xcur, outX, lng, lnb);
}
